// MultiHeadDifAttention_32031866093742
// MI455X (gfx1250) — compile-verified, matrix operand order checked by reading, not run on hardware
//
#include <hip/hip_runtime.h>
#include <hip/hip_bf16.h>

// ---------------------------------------------------------------------------
// Problem constants (from reference)
// ---------------------------------------------------------------------------
#define BB 2
#define SS 2048
#define HH 8
#define DD 128
#define EPSV 1e-5f
#define LAMBDA_INIT 0.5f

typedef __attribute__((ext_vector_type(16))) _Float16 v16h;
typedef __attribute__((ext_vector_type(8)))  float    v8f;
typedef __attribute__((ext_vector_type(4)))  _Float16 v4h;
typedef __attribute__((ext_vector_type(4)))  float    v4f;
typedef __attribute__((ext_vector_type(4)))  int      v4i;

union FragH { v16h v; _Float16 h[16]; };
union AccF  { v8f  v; float    f[8];  };

// ---------------------------------------------------------------------------
// CDNA5 async global->LDS staging (ASYNCcnt path), guarded so that a
// toolchain without the builtin falls back to the proven synchronous path.
// Builtin signature (from compiler diagnostic): (int4 AS1*, int4 AS3*, Ii, Ii)
// ---------------------------------------------------------------------------
#if defined(__has_builtin)
#if __has_builtin(__builtin_amdgcn_global_load_async_to_lds_b128)
#define USE_ASYNC_LDS 1
#else
#define USE_ASYNC_LDS 0
#endif
#else
#define USE_ASYNC_LDS 0
#endif

#if USE_ASYNC_LDS
typedef __attribute__((address_space(1))) v4i g_v4i;
typedef __attribute__((address_space(3))) v4i l_v4i;

__device__ inline void async_ld16(const _Float16* g, _Float16* l) {
    __builtin_amdgcn_global_load_async_to_lds_b128((g_v4i*)g, (l_v4i*)l, 0, 0);
}
#endif

template <int N>
__device__ inline void wait_asyncN() {
#if USE_ASYNC_LDS
#if __has_builtin(__builtin_amdgcn_s_wait_asynccnt)
    __builtin_amdgcn_s_wait_asynccnt(N);
#else
    asm volatile("s_wait_asynccnt %0" :: "i"(N) : "memory");
#endif
#endif
}

// ---------------------------------------------------------------------------
// WMMA fragment load from a row-major [16 x >=32] LDS tile (stride ld halves).
// A-layout (16-bit, 16x32): lane<16 holds row=lane, K = 0..7 and 16..23;
// lane>=16 holds row=lane-16, K = 8..15 and 24..31.  B uses mirrored layout
// from an N-major (transposed) tile, so the same loader serves both.
// ---------------------------------------------------------------------------
__device__ inline v16h ld_frag(const _Float16* __restrict__ base, int ld, int lane) {
    const int r  = lane & 15;
    const int kh = (lane >> 4) << 3;           // 0 or 8
    const _Float16* p0 = base + r * ld + kh;
    const _Float16* p1 = p0 + 16;
    FragH f;
#pragma unroll
    for (int i = 0; i < 8; ++i) f.h[i]     = p0[i];
#pragma unroll
    for (int i = 0; i < 8; ++i) f.h[8 + i] = p1[i];
    return f.v;
}

// ---------------------------------------------------------------------------
// Generic batched WMMA GEMM: C[M,N] = scale * A[M,K] * B + bias
//   B_NK  : B supplied as N x K row-major (i.e. B^T; used for Q*K^T)
//   OUT16 : store C as f16 (else f32)
//   A_F32 : A supplied as f32, converted to f16 while staging into LDS
// Block tile 128x128, K-step 32, 256 threads = 8 waves (4 along M, 2 along N),
// each wave computes a 32x64 subtile via 8 x v_wmma_f32_16x16x32_f16.
// Double-buffered LDS: tile t+1 is staged (async DMA where possible) before
// computing tile t, with s_wait_asynccnt <N_next> to drain only the old batch.
// All dims must be multiples of the tile sizes (true for every launch below).
// ---------------------------------------------------------------------------
#define BM 128
#define BN 128
#define BKK 32
#define LDW 40   // padded LDS row stride in halves (80B rows -> 16B aligned cols)
#define TSZ (BM * LDW)

template <bool B_NK, bool OUT16, bool A_F32>
__global__ __launch_bounds__(256)
void gemm_wmma(const void* __restrict__ Ap, const _Float16* __restrict__ Bmat,
               void* __restrict__ Cp, const float* __restrict__ bias, float scale,
               int M, int N, int K, int lda, int ldb, int ldc,
               long long sA, long long sB, long long sC)
{
    __shared__ __align__(16) _Float16 As[2 * TSZ];
    __shared__ __align__(16) _Float16 Bs[2 * TSZ];

    const int tid  = threadIdx.x;
    const int lane = tid & 31;
    const int wave = tid >> 5;
    const int wm   = wave & 3;      // wave row group (32 rows each)
    const int wn   = wave >> 2;     // wave col group (64 cols each)
    const int bz   = blockIdx.z;
    const int rowBase = blockIdx.y * BM;
    const int colBase = blockIdx.x * BN;

    const _Float16* A16 = (const _Float16*)Ap;
    const float*    A32 = (const float*)Ap;
    if (A_F32) A32 += (size_t)bz * sA; else A16 += (size_t)bz * sA;
    const _Float16* Bp = Bmat + (size_t)bz * sB;

    // # of async instructions a wave issues per stage (immediate for waits)
    constexpr int ASYNC_N = (A_F32 ? 0 : 2) + (B_NK ? 2 : 0);

    // ---- stage one K-tile (A and B) into LDS buffer `buf` ----
    auto stage = [&](int buf, int k0) {
        _Float16* Ad = &As[buf * TSZ];
        _Float16* Bd = &Bs[buf * TSZ];
        // A tile: 128 x 32 halves
        if (!A_F32) {
#if USE_ASYNC_LDS
#pragma unroll
            for (int t = 0; t < 2; ++t) {
                int c   = tid + t * 256;
                int r   = c >> 2;            // 0..127
                int col = (c & 3) << 3;      // 0,8,16,24 (halves)
                async_ld16(A16 + (size_t)(rowBase + r) * lda + k0 + col,
                           &Ad[r * LDW + col]);
            }
#else
#pragma unroll
            for (int t = 0; t < 4; ++t) {
                int c   = tid + t * 256;
                int r   = c >> 3;            // 0..127
                int col = (c & 7) << 2;      // 0,4,...,28
                v4h d = *(const v4h*)(A16 + (size_t)(rowBase + r) * lda + k0 + col);
                *(v4h*)&Ad[r * LDW + col] = d;
            }
#endif
        } else {
#pragma unroll
            for (int t = 0; t < 4; ++t) {
                int c   = tid + t * 256;
                int r   = c >> 3;
                int col = (c & 7) << 2;
                v4f d = *(const v4f*)(A32 + (size_t)(rowBase + r) * lda + k0 + col);
                v4h h;
                h[0] = (_Float16)d[0]; h[1] = (_Float16)d[1];
                h[2] = (_Float16)d[2]; h[3] = (_Float16)d[3];
                *(v4h*)&Ad[r * LDW + col] = h;
            }
        }
        // B tile into N-major layout Bd[n][k]
        if (B_NK) {
#if USE_ASYNC_LDS
#pragma unroll
            for (int t = 0; t < 2; ++t) {
                int c   = tid + t * 256;
                int r   = c >> 2;
                int col = (c & 3) << 3;
                async_ld16(Bp + (size_t)(colBase + r) * ldb + k0 + col,
                           &Bd[r * LDW + col]);
            }
#else
#pragma unroll
            for (int t = 0; t < 4; ++t) {
                int c   = tid + t * 256;
                int r   = c >> 3;
                int col = (c & 7) << 2;
                v4h d = *(const v4h*)(Bp + (size_t)(colBase + r) * ldb + k0 + col);
                *(v4h*)&Bd[r * LDW + col] = d;
            }
#endif
        } else {
#pragma unroll
            for (int t = 0; t < 4; ++t) {
                int c  = tid + t * 256;
                int kk = c >> 5;             // 0..31
                int nh = (c & 31) << 2;      // 0,4,...,124
                v4h d = *(const v4h*)(Bp + (size_t)(k0 + kk) * ldb + colBase + nh);
                Bd[(nh + 0) * LDW + kk] = d[0];
                Bd[(nh + 1) * LDW + kk] = d[1];
                Bd[(nh + 2) * LDW + kk] = d[2];
                Bd[(nh + 3) * LDW + kk] = d[3];
            }
        }
    };

    v8f acc[2][4] = {};
    const int nk = K / BKK;

    stage(0, 0);                               // prologue: fill buffer 0
    for (int t = 0; t < nk; ++t) {
        const int cur = t & 1;
        const bool hasNext = (t + 1 < nk);
        if (hasNext) stage(cur ^ 1, (t + 1) * BKK);   // DMA overlaps compute
        if (ASYNC_N > 0) {
            if (hasNext) wait_asyncN<ASYNC_N>();      // only next batch pending
            else         wait_asyncN<0>();
        }
        __syncthreads();                        // cur tile visible block-wide

        const _Float16* Ac = &As[cur * TSZ];
        const _Float16* Bc = &Bs[cur * TSZ];
        v16h af[2], bf[4];
#pragma unroll
        for (int i = 0; i < 2; ++i) af[i] = ld_frag(&Ac[(wm * 32 + i * 16) * LDW], LDW, lane);
#pragma unroll
        for (int j = 0; j < 4; ++j) bf[j] = ld_frag(&Bc[(wn * 64 + j * 16) * LDW], LDW, lane);

#pragma unroll
        for (int i = 0; i < 2; ++i)
#pragma unroll
            for (int j = 0; j < 4; ++j)
                acc[i][j] = __builtin_amdgcn_wmma_f32_16x16x32_f16(
                    false, af[i], false, bf[j], (short)0, acc[i][j], false, false);
        __syncthreads();                        // protect cur buffer for reuse
    }

    // ---- epilogue: C/D layout -> lane<16: row=e,col=lane ; lane>=16: row=e+8 ----
    float*    C32 = (float*)Cp;
    _Float16* C16 = (_Float16*)Cp;
    if (OUT16) C16 += (size_t)bz * sC; else C32 += (size_t)bz * sC;
    const int rlo = lane & 15;
    const int rhi = (lane >> 4) << 3;

    float bvals[4];
#pragma unroll
    for (int j = 0; j < 4; ++j)
        bvals[j] = bias ? bias[colBase + wn * 64 + j * 16 + rlo] : 0.0f;

#pragma unroll
    for (int i = 0; i < 2; ++i) {
#pragma unroll
        for (int j = 0; j < 4; ++j) {
            AccF a; a.v = acc[i][j];
            const int col = colBase + wn * 64 + j * 16 + rlo;
#pragma unroll
            for (int e = 0; e < 8; ++e) {
                int row = rowBase + wm * 32 + i * 16 + e + rhi;
                float v = a.f[e] * scale + bvals[j];
                if (OUT16) C16[(size_t)row * ldc + col] = (_Float16)v;
                else       C32[(size_t)row * ldc + col] = v;
            }
        }
    }
}

// ---------------------------------------------------------------------------
// f32 -> f16 conversion
// ---------------------------------------------------------------------------
__global__ __launch_bounds__(256)
void f32_to_f16_kernel(const float* __restrict__ in, _Float16* __restrict__ out, long long n) {
    long long i = (long long)blockIdx.x * blockDim.x + threadIdx.x;
    long long st = (long long)gridDim.x * blockDim.x;
    for (; i < n; i += st) out[i] = (_Float16)in[i];
}

// ---------------------------------------------------------------------------
// Row stats for the two logit planes: per row -> (m1, 1/l1, m2, 1/l2)
// ---------------------------------------------------------------------------
__global__ __launch_bounds__(256)
void rowstats_kernel(const float* __restrict__ l1, const float* __restrict__ l2,
                     float* __restrict__ st, int Sdim) {
    const int row = blockIdx.x;
    const int tid = threadIdx.x;
    const float* r1 = l1 + (size_t)row * Sdim;
    const float* r2 = l2 + (size_t)row * Sdim;
    __shared__ float red[256];

    float m1 = -1e30f, m2 = -1e30f;
    for (int j = tid; j < Sdim; j += 256) {
        m1 = fmaxf(m1, r1[j]);
        m2 = fmaxf(m2, r2[j]);
    }
    red[tid] = m1; __syncthreads();
    for (int o = 128; o > 0; o >>= 1) { if (tid < o) red[tid] = fmaxf(red[tid], red[tid + o]); __syncthreads(); }
    m1 = red[0]; __syncthreads();
    red[tid] = m2; __syncthreads();
    for (int o = 128; o > 0; o >>= 1) { if (tid < o) red[tid] = fmaxf(red[tid], red[tid + o]); __syncthreads(); }
    m2 = red[0]; __syncthreads();

    float s1 = 0.f, s2 = 0.f;
    for (int j = tid; j < Sdim; j += 256) {
        s1 += __expf(r1[j] - m1);
        s2 += __expf(r2[j] - m2);
    }
    red[tid] = s1; __syncthreads();
    for (int o = 128; o > 0; o >>= 1) { if (tid < o) red[tid] += red[tid + o]; __syncthreads(); }
    s1 = red[0]; __syncthreads();
    red[tid] = s2; __syncthreads();
    for (int o = 128; o > 0; o >>= 1) { if (tid < o) red[tid] += red[tid + o]; __syncthreads(); }
    s2 = red[0];

    if (tid == 0) {
        st[(size_t)row * 4 + 0] = m1;
        st[(size_t)row * 4 + 1] = 1.0f / s1;
        st[(size_t)row * 4 + 2] = m2;
        st[(size_t)row * 4 + 3] = 1.0f / s2;
    }
}

// ---------------------------------------------------------------------------
// attn = exp(s1-m1)/l1 - lam * exp(s2-m2)/l2   (in place over s1 plane)
// ---------------------------------------------------------------------------
__global__ __launch_bounds__(256)
void attn_finalize_kernel(float* __restrict__ attn, const float* __restrict__ l2,
                          const float* __restrict__ st, const float* __restrict__ lamp,
                          long long total, int Sdim) {
    const float lam = lamp[0];
    long long i  = (long long)blockIdx.x * blockDim.x + threadIdx.x;
    long long sp = (long long)gridDim.x * blockDim.x;
    for (; i < total; i += sp) {
        long long row = i / Sdim;
        const float m1  = st[row * 4 + 0];
        const float il1 = st[row * 4 + 1];
        const float m2  = st[row * 4 + 2];
        const float il2 = st[row * 4 + 3];
        attn[i] = __expf(attn[i] - m1) * il1 - lam * __expf(l2[i] - m2) * il2;
    }
}

// ---------------------------------------------------------------------------
// GroupNorm stats over the reference's relabeled (B,H,S,D) view of the
// transposed out tensor: group (b,g) covers flat positions p = g*S*D + j of
// y[b] where y[(s*H+h)*D+d] = o[b][h][s][d].
// ---------------------------------------------------------------------------
__global__ __launch_bounds__(256)
void gn_stats_kernel(const float* __restrict__ o, float* __restrict__ mv) {
    const int b = blockIdx.x / HH;
    const int g = blockIdx.x % HH;
    const int tid = threadIdx.x;
    __shared__ float rs[256], rq[256];
    float s = 0.f, q = 0.f;
    const int SD = SS * DD;
    for (int j = tid; j < SD; j += 256) {
        int p  = g * SD + j;
        int dy = p % DD;
        int hy = (p / DD) % HH;
        int sy = p / (DD * HH);
        float v = o[(((size_t)b * HH + hy) * SS + sy) * DD + dy];
        s += v; q += v * v;
    }
    rs[tid] = s; rq[tid] = q; __syncthreads();
    for (int oo = 128; oo > 0; oo >>= 1) {
        if (tid < oo) { rs[tid] += rs[tid + oo]; rq[tid] += rq[tid + oo]; }
        __syncthreads();
    }
    if (tid == 0) {
        float mu  = rs[0] / (float)SD;
        float var = rq[0] / (float)SD - mu * mu;
        mv[blockIdx.x * 2 + 0] = mu;
        mv[blockIdx.x * 2 + 1] = rsqrtf(var + EPSV);
    }
}

// ---------------------------------------------------------------------------
// Apply GN + affine + (1-LAMBDA_INIT), emit f16 in the final (B,S,H*D) layout.
// Output index n = ((b*S+s)*H+g)*D+d  maps to y[b] position p=g*S*D+s*D+d,
// which maps back to o[b][hy][sy][dy].
// ---------------------------------------------------------------------------
__global__ __launch_bounds__(256)
void gn_apply_kernel(const float* __restrict__ o, const float* __restrict__ mv,
                     const float* __restrict__ gw, const float* __restrict__ gb,
                     _Float16* __restrict__ z, long long total) {
    long long i  = (long long)blockIdx.x * blockDim.x + threadIdx.x;
    long long sp = (long long)gridDim.x * blockDim.x;
    const long long SHD = (long long)SS * HH * DD;
    for (; i < total; i += sp) {
        long long b   = i / SHD;
        long long rem = i % SHD;
        int s = (int)(rem / (HH * DD));
        int g = (int)((rem / DD) % HH);
        int d = (int)(rem % DD);
        long long p = (long long)g * SS * DD + (long long)s * DD + d;
        int dy = (int)(p % DD);
        int hy = (int)((p / DD) % HH);
        int sy = (int)(p / (DD * HH));
        float v   = o[((b * HH + hy) * SS + sy) * DD + dy];
        float mu  = mv[(b * HH + g) * 2 + 0];
        float rst = mv[(b * HH + g) * 2 + 1];
        float r = ((v - mu) * rst * gw[g] + gb[g]) * (1.0f - LAMBDA_INIT);
        z[i] = (_Float16)r;
    }
}

// ---------------------------------------------------------------------------
// Host-side orchestration
// ---------------------------------------------------------------------------
extern "C" void kernel_launch(void* const* d_in, const int* in_sizes, int n_in,
                              void* d_out, int out_size, void* d_ws, size_t ws_size,
                              hipStream_t stream) {
    (void)in_sizes; (void)n_in; (void)out_size; (void)ws_size;

    const float* x   = (const float*)d_in[0];
    const float* Wq  = (const float*)d_in[1];
    const float* bq  = (const float*)d_in[2];
    const float* Wk  = (const float*)d_in[3];
    const float* bk  = (const float*)d_in[4];
    const float* Wv  = (const float*)d_in[5];
    const float* bv  = (const float*)d_in[6];
    const float* Wo  = (const float*)d_in[7];
    const float* bo  = (const float*)d_in[8];
    const float* gnw = (const float*)d_in[9];
    const float* gnb = (const float*)d_in[10];
    const float* lam = (const float*)d_in[11];

    const int M = BB * SS;                       // 4096
    const long long nAttn = (long long)BB * HH * SS * SS;

    // workspace carve-out
    char* w = (char*)d_ws;
    auto alloc = [&](size_t bytes) { void* p = (void*)w; w += (bytes + 255) & ~(size_t)255; return p; };
    _Float16* xh  = (_Float16*)alloc((size_t)M * DD * 2);
    _Float16* Wqh = (_Float16*)alloc((size_t)DD * 2 * HH * DD * 2);
    _Float16* Wkh = (_Float16*)alloc((size_t)DD * 2 * HH * DD * 2);
    _Float16* Wvh = (_Float16*)alloc((size_t)DD * HH * DD * 2);
    _Float16* Woh = (_Float16*)alloc((size_t)HH * DD * DD * 2);
    _Float16* qh  = (_Float16*)alloc((size_t)M * 2 * HH * DD * 2);
    _Float16* kh  = (_Float16*)alloc((size_t)M * 2 * HH * DD * 2);
    _Float16* vh  = (_Float16*)alloc((size_t)M * HH * DD * 2);
    float*    lg2 = (float*)   alloc((size_t)nAttn * 4);
    float*    ost = (float*)   alloc((size_t)BB * HH * SS * DD * 4);   // pre-GN out
    float*    st  = (float*)   alloc((size_t)BB * HH * SS * 4 * 4);    // row stats
    float*    mv  = (float*)   alloc((size_t)BB * HH * 2 * 4);         // mu,rstd
    _Float16* zh  = (_Float16*)alloc((size_t)M * HH * DD * 2);         // GN'd f16

    float*    outp = (float*)d_out;                                    // (B,S,D)
    float*    attn = outp + (size_t)BB * SS * DD;                      // (B,H,S,S)

    const dim3 blk(256);
    const float scl = 0.08838834764831845f;      // 1/sqrt(128)

    // 1) f32 -> f16 conversions
    f32_to_f16_kernel<<<1024, blk, 0, stream>>>(x,  xh,  (long long)M * DD);
    f32_to_f16_kernel<<<1024, blk, 0, stream>>>(Wq, Wqh, (long long)DD * 2 * HH * DD);
    f32_to_f16_kernel<<<1024, blk, 0, stream>>>(Wk, Wkh, (long long)DD * 2 * HH * DD);
    f32_to_f16_kernel<<<1024, blk, 0, stream>>>(Wv, Wvh, (long long)DD * HH * DD);
    f32_to_f16_kernel<<<1024, blk, 0, stream>>>(Wo, Woh, (long long)HH * DD * DD);

    // 2) QKV projections: (4096 x 128) * (128 x N), f16 out + bias
    gemm_wmma<false, true, false><<<dim3(2048 / BN, M / BM, 1), blk, 0, stream>>>(
        xh, Wqh, qh, bq, 1.0f, M, 2048, 128, 128, 2048, 2048, 0, 0, 0);
    gemm_wmma<false, true, false><<<dim3(2048 / BN, M / BM, 1), blk, 0, stream>>>(
        xh, Wkh, kh, bk, 1.0f, M, 2048, 128, 128, 2048, 2048, 0, 0, 0);
    gemm_wmma<false, true, false><<<dim3(1024 / BN, M / BM, 1), blk, 0, stream>>>(
        xh, Wvh, vh, bv, 1.0f, M, 1024, 128, 128, 1024, 1024, 0, 0, 0);

    // 3) Scores: 16 batches of (2048 x 128) * (128 x 2048), B given as K-rows
    //    (N x K) => contiguous Kᵀ staging via async DMA. Scaled logits in f32.
    const long long sQK = (long long)SS * 2 * DD;      // per-(b,h) stride in q/k
    const long long sAT = (long long)SS * SS;
    gemm_wmma<true, false, false><<<dim3(SS / BN, SS / BM, BB * HH), blk, 0, stream>>>(
        qh, kh, attn, nullptr, scl, SS, SS, DD, 2 * DD, 2 * DD, SS, sQK, sQK, sAT);
    gemm_wmma<true, false, false><<<dim3(SS / BN, SS / BM, BB * HH), blk, 0, stream>>>(
        qh + DD, kh + DD, lg2, nullptr, scl, SS, SS, DD, 2 * DD, 2 * DD, SS, sQK, sQK, sAT);

    // 4) softmax stats + differential combine (in place into d_out attn plane)
    rowstats_kernel<<<BB * HH * SS, blk, 0, stream>>>(attn, lg2, st, SS);
    attn_finalize_kernel<<<32768, blk, 0, stream>>>(attn, lg2, st, lam, nAttn, SS);

    // 5) out = attn @ v : 16 batches of (2048 x 2048) * (2048 x 128),
    //    A read as f32 and converted to f16 while staging into LDS.
    const long long sV = (long long)SS * DD;
    gemm_wmma<false, false, true><<<dim3(DD / BN, SS / BM, BB * HH), blk, 0, stream>>>(
        attn, vh, ost, nullptr, 1.0f, SS, DD, SS, SS, DD, DD, sAT, sV, sV);

    // 6) GroupNorm over relabeled view + affine + (1-lambda_init), f16 out
    gn_stats_kernel<<<BB * HH, blk, 0, stream>>>(ost, mv);
    gn_apply_kernel<<<4096, blk, 0, stream>>>(ost, mv, gnw, gnb, zh, (long long)M * HH * DD);

    // 7) Final projection: (4096 x 1024) * (1024 x 128) + bo -> d_out
    gemm_wmma<false, false, false><<<dim3(DD / BN, M / BM, 1), blk, 0, stream>>>(
        zh, Woh, outp, bo, 1.0f, M, DD, HH * DD, HH * DD, DD, DD, 0, 0, 0);
}
